// MultiHeadAttention_9612136808854
// MI455X (gfx1250) — compile-verified, matrix operand order checked by reading, not run on hardware
//
#include <hip/hip_runtime.h>

typedef __attribute__((ext_vector_type(16))) __bf16 v16bf;
typedef __attribute__((ext_vector_type(8)))  __bf16 v8bf;
typedef __attribute__((ext_vector_type(8)))  float  v8f;

#define BATCH 4
#define SEQ   2048
#define NEMB  1024
#define NHEAD 16
#define HDIM  64
#define MROWS (BATCH * SEQ)   /* 8192 */

// ---- gfx1250 async global->LDS path (guarded: falls back to sync staging) ----
#if defined(__has_builtin)
#  if __has_builtin(__builtin_amdgcn_global_load_async_to_lds_b128) && \
      __has_builtin(__builtin_amdgcn_s_wait_asynccnt)
#    define USE_ASYNC_LDS 1
#  endif
#endif
#ifndef USE_ASYNC_LDS
#  define USE_ASYNC_LDS 0
#endif

#if USE_ASYNC_LDS
typedef __attribute__((ext_vector_type(4))) int v4i_t;
static __device__ __forceinline__ void async_cp16(const __bf16* g, __bf16* l) {
  // prototype (from hipcc diagnostic): (int4*, int4*, imm offset, imm cpol)
  __builtin_amdgcn_global_load_async_to_lds_b128((v4i_t*)(void*)g, (v4i_t*)(void*)l, 0, 0);
}
#endif

// ---------- helpers ----------
static __device__ __forceinline__ v8bf ldb8(const __bf16* p) {
  return *reinterpret_cast<const v8bf*>(p);
}
static __device__ __forceinline__ v16bf cat8(v8bf a, v8bf b) {
  return __builtin_shufflevector(a, b, 0,1,2,3,4,5,6,7,8,9,10,11,12,13,14,15);
}
static __device__ __forceinline__ v8f wmma_bf16(v16bf a, v16bf b, v8f c) {
  // (neg_a, A, neg_b, B, c_mod, C, reuse_a, reuse_b)
  return __builtin_amdgcn_wmma_f32_16x16x32_bf16(false, a, false, b, (short)0, c, false, false);
}

// ---------- f32 -> bf16 conversion (x8 vectorized) ----------
__global__ __launch_bounds__(256) void cvt_f32_bf16(const float* __restrict__ src,
                                                    __bf16* __restrict__ dst, int n8) {
  int i = blockIdx.x * blockDim.x + threadIdx.x;
  if (i >= n8) return;
  const float4* s = reinterpret_cast<const float4*>(src) + (size_t)i * 2;
  float4 f0 = s[0], f1 = s[1];
  v8bf o;
  o[0] = (__bf16)f0.x; o[1] = (__bf16)f0.y; o[2] = (__bf16)f0.z; o[3] = (__bf16)f0.w;
  o[4] = (__bf16)f1.x; o[5] = (__bf16)f1.y; o[6] = (__bf16)f1.z; o[7] = (__bf16)f1.w;
  reinterpret_cast<v8bf*>(dst)[i] = o;
}

// ---------- projection GEMM: Y = A(bf16 [M,1024]) @ W(bf16 [N,1024])^T ----------
// mode 0: store bf16 scattered to [B,H,T,D]    (Q, K)
// mode 1: store bf16 scattered to [B,H,D,T]    (V transposed; packed b128 stores)
// mode 2: store f32 + bias to [M, 1024]        (final output)
__global__ __launch_bounds__(256) void proj_gemm(const __bf16* __restrict__ A,
                                                 const __bf16* __restrict__ W,
                                                 void* __restrict__ outp,
                                                 const float* __restrict__ bias,
                                                 int mode) {
  __shared__ alignas(16) __bf16 Bs[2][64 * 32];  // double-buffered W tile
  const int tid  = threadIdx.x;
  const int w    = tid >> 5;
  const int lane = tid & 31;
  const int Nb   = blockIdx.x * 64;
  const int Mb   = blockIdx.y * 128;
  const int lr   = lane & 15;
  const int grp  = lane >> 4;
  const int off  = grp ? 8 : 0;

  const __bf16* Arow = A + (size_t)(Mb + w * 16 + lr) * NEMB;

  // per-thread 16B slice of the W tile
  const int r  = tid >> 2, c4 = tid & 3;
  const __bf16* wsrc = W + (size_t)(Nb + r) * NEMB + c4 * 8;
  const int     loff = r * 32 + c4 * 8;

#if USE_ASYNC_LDS
  async_cp16(wsrc, &Bs[0][loff]);                // prologue: tile 0 in flight
#endif

  v8f acc[4] = {};

  for (int kb = 0; kb < NEMB; kb += 32) {
    const int buf = (kb >> 5) & 1;
#if USE_ASYNC_LDS
    // issue next tile into the other buffer (tail issues a harmless dummy so
    // the in-flight count is always 2 and the wait immediate stays constant)
    const int knext = (kb + 32 < NEMB) ? (kb + 32) : 0;
    async_cp16(wsrc + knext, &Bs[buf ^ 1][loff]);
    __builtin_amdgcn_s_wait_asynccnt(1);         // current tile's fill complete
    __syncthreads();
#else
    {
      v8bf v = ldb8(wsrc + kb);
      *reinterpret_cast<v8bf*>(&Bs[buf][loff]) = v;
    }
    __syncthreads();
#endif

    v16bf af = cat8(ldb8(Arow + kb + off), ldb8(Arow + kb + 16 + off));
    v16bf bw[4];
#pragma unroll
    for (int s = 0; s < 4; ++s) {
      const __bf16* brow = &Bs[buf][(s * 16 + lr) * 32];
      bw[s] = cat8(ldb8(brow + off), ldb8(brow + 16 + off));
    }
#pragma unroll
    for (int s = 0; s < 4; ++s) acc[s] = wmma_bf16(af, bw[s], acc[s]);

    __syncthreads();                             // readers done before next fill
  }

  // C layout: lane (grp, lr) holds reg i -> (M = i + 8*grp, N = lr)
  const int m0 = Mb + w * 16 + 8 * grp;
  if (mode == 2) {
    float* out = (float*)outp;
#pragma unroll
    for (int s = 0; s < 4; ++s) {
      int ng = Nb + s * 16 + lr;
      float bv = bias[ng];
#pragma unroll
      for (int i = 0; i < 8; ++i)
        out[(size_t)(m0 + i) * NEMB + ng] = acc[s][i] + bv;
    }
  } else if (mode == 0) {
    __bf16* out = (__bf16*)outp;
#pragma unroll
    for (int s = 0; s < 4; ++s) {
      int ng = Nb + s * 16 + lr;
      int h = ng >> 6, d = ng & 63;
#pragma unroll
      for (int i = 0; i < 8; ++i) {
        int mg = m0 + i;
        int b = mg >> 11, t = mg & 2047;
        out[((size_t)((b * NHEAD + h) * SEQ + t)) * HDIM + d] = (__bf16)acc[s][i];
      }
    }
  } else {  // mode 1: V^T [B,H,D,T] -- 8 consecutive t pack into one b128
    __bf16* out = (__bf16*)outp;
    int b = m0 >> 11, t0 = m0 & 2047;
#pragma unroll
    for (int s = 0; s < 4; ++s) {
      int ng = Nb + s * 16 + lr;
      int h = ng >> 6, d = ng & 63;
      v8bf pk;
#pragma unroll
      for (int i = 0; i < 8; ++i) pk[i] = (__bf16)acc[s][i];
      *reinterpret_cast<v8bf*>(&out[((size_t)((b * NHEAD + h) * HDIM + d)) * SEQ + t0]) = pk;
    }
  }
}

// ---------- flash attention: one wave per (b,h, 16-query tile) ----------
__global__ __launch_bounds__(256) void attn_kernel(const __bf16* __restrict__ Q,
                                                   const __bf16* __restrict__ Kp,
                                                   const __bf16* __restrict__ Vt,
                                                   __bf16* __restrict__ Oa) {
  __shared__ alignas(16) __bf16 Ps[8][16 * 32];  // per-wave P tile (1KB each)
  const int tid  = threadIdx.x;
  const int w    = tid >> 5;
  const int lane = tid & 31;
  const int wid  = blockIdx.x * 8 + w;
  const int bh   = wid >> 7;                     // b*16+h
  const int qt   = wid & 127;                    // query tile
  const int lr   = lane & 15;
  const int grp  = lane >> 4;
  const int off  = grp ? 8 : 0;
  const float SCALE = 0.125f;                    // 1/sqrt(64)

  const __bf16* Qb = Q  + (size_t)bh * SEQ * HDIM;
  const __bf16* Kb = Kp + (size_t)bh * SEQ * HDIM;
  const __bf16* Vb = Vt + (size_t)bh * HDIM * SEQ;

  // Q fragments (rows qt*16 .. +15, two 32-wide d-chunks), kept in registers
  v16bf aQ[2];
#pragma unroll
  for (int dc = 0; dc < 2; ++dc) {
    const __bf16* qrow = Qb + (size_t)(qt * 16 + lr) * HDIM + dc * 32;
    aQ[dc] = cat8(ldb8(qrow + off), ldb8(qrow + 16 + off));
  }

  v8f o[4] = {};
  float mrow[8], rsum[8];
#pragma unroll
  for (int i = 0; i < 8; ++i) { mrow[i] = -1.0e30f; rsum[i] = 0.0f; }

  const int kend = qt * 16 + 16;                 // exclusive causal key bound
  for (int kb = 0; kb < kend; kb += 32) {
    // prefetch next key/value block toward the WGP while this one computes
    const int kpf = (kb + 32 < kend) ? (kb + 32) : kb;
    __builtin_prefetch(Kb + (size_t)(kpf + lane) * HDIM, 0, 3);
    __builtin_prefetch(Vb + (size_t)(lane * 2) * SEQ + kpf, 0, 3);
    __builtin_prefetch(Vb + (size_t)(lane * 2 + 1) * SEQ + kpf, 0, 3);

    // load all 4 K fragments, then a dense WMMA burst for S = Q K^T
    v16bf bk[2][2];
#pragma unroll
    for (int kt = 0; kt < 2; ++kt) {
      const __bf16* krow = Kb + (size_t)(kb + kt * 16 + lr) * HDIM;
#pragma unroll
      for (int dc = 0; dc < 2; ++dc)
        bk[kt][dc] = cat8(ldb8(krow + dc * 32 + off), ldb8(krow + dc * 32 + 16 + off));
    }
    v8f sv[2] = {};
#pragma unroll
    for (int kt = 0; kt < 2; ++kt)
#pragma unroll
      for (int dc = 0; dc < 2; ++dc)
        sv[kt] = wmma_bf16(aQ[dc], bk[kt][dc], sv[kt]);

    // scale, causal mask, online softmax (16-lane butterfly reductions)
    const int k0 = kb + lr, k1 = kb + 16 + lr;
#pragma unroll
    for (int i = 0; i < 8; ++i) {
      const int tg = qt * 16 + i + 8 * grp;
      float v0 = sv[0][i] * SCALE; if (k0 > tg) v0 = -1.0e30f;
      float v1 = sv[1][i] * SCALE; if (k1 > tg) v1 = -1.0e30f;
      float mx = fmaxf(v0, v1);
#pragma unroll
      for (int d = 1; d < 16; d <<= 1) mx = fmaxf(mx, __shfl_xor(mx, d, 32));
      float mnew  = fmaxf(mrow[i], mx);
      float alpha = __expf(mrow[i] - mnew);
      mrow[i] = mnew;
      float p0 = __expf(v0 - mnew), p1 = __expf(v1 - mnew);
      float rs = p0 + p1;
#pragma unroll
      for (int d = 1; d < 16; d <<= 1) rs += __shfl_xor(rs, d, 32);
      rsum[i] = rsum[i] * alpha + rs;
#pragma unroll
      for (int j = 0; j < 4; ++j) o[j][i] *= alpha;
      const int rr = i + 8 * grp;                // C-layout -> LDS row-major P
      Ps[w][rr * 32 + lr]      = (__bf16)p0;
      Ps[w][rr * 32 + 16 + lr] = (__bf16)p1;
    }

    // P fragment from per-wave LDS (same-wave DS ops are in-order)
    const __bf16* prow = &Ps[w][lr * 32];
    v16bf aP = cat8(ldb8(prow + off), ldb8(prow + 16 + off));

    // load all 4 V fragments, then a dense WMMA burst for O += P x V
    v16bf bv[4];
#pragma unroll
    for (int j = 0; j < 4; ++j) {
      const __bf16* vrow = Vb + (size_t)(j * 16 + lr) * SEQ + kb;
      bv[j] = cat8(ldb8(vrow + off), ldb8(vrow + 16 + off));
    }
#pragma unroll
    for (int j = 0; j < 4; ++j) o[j] = wmma_bf16(aP, bv[j], o[j]);
  }

  // epilogue: normalize and scatter to [B, T, C] (heads re-interleaved), bf16
  const int b = bh >> 4, h = bh & 15;
#pragma unroll
  for (int j = 0; j < 4; ++j) {
    const int col = h * 64 + j * 16 + lr;
#pragma unroll
    for (int i = 0; i < 8; ++i) {
      const int tg = qt * 16 + i + 8 * grp;
      Oa[(size_t)(b * SEQ + tg) * NEMB + col] = (__bf16)(o[j][i] / rsum[i]);
    }
  }
}

// ---------- launcher ----------
extern "C" void kernel_launch(void* const* d_in, const int* in_sizes, int n_in,
                              void* d_out, int out_size, void* d_ws, size_t ws_size,
                              hipStream_t stream) {
  (void)in_sizes; (void)n_in; (void)out_size; (void)ws_size;
  const float* x  = (const float*)d_in[0];
  const float* Wk = (const float*)d_in[1];
  const float* Wq = (const float*)d_in[2];
  const float* Wv = (const float*)d_in[3];
  const float* Wp = (const float*)d_in[4];
  const float* bp = (const float*)d_in[5];

  const size_t EL  = (size_t)MROWS * NEMB;       // 8388608
  const size_t WEL = (size_t)NEMB * NEMB;        // 1048576
  char* p = (char*)d_ws;
  __bf16* xb  = (__bf16*)p; p += EL * 2;
  __bf16* wkb = (__bf16*)p; p += WEL * 2;
  __bf16* wqb = (__bf16*)p; p += WEL * 2;
  __bf16* wvb = (__bf16*)p; p += WEL * 2;
  __bf16* wpb = (__bf16*)p; p += WEL * 2;
  __bf16* qb  = (__bf16*)p; p += EL * 2;         // [B,H,T,D]
  __bf16* kbf = (__bf16*)p; p += EL * 2;         // [B,H,T,D]
  __bf16* vtb = (__bf16*)p; p += EL * 2;         // [B,H,D,T]
  __bf16* ab  = (__bf16*)p; p += EL * 2;         // [M, C] attention output

  // conversions
  cvt_f32_bf16<<<(int)((EL / 8 + 255) / 256), 256, 0, stream>>>(x, xb, (int)(EL / 8));
  cvt_f32_bf16<<<(int)((WEL / 8 + 255) / 256), 256, 0, stream>>>(Wk, wkb, (int)(WEL / 8));
  cvt_f32_bf16<<<(int)((WEL / 8 + 255) / 256), 256, 0, stream>>>(Wq, wqb, (int)(WEL / 8));
  cvt_f32_bf16<<<(int)((WEL / 8 + 255) / 256), 256, 0, stream>>>(Wv, wvb, (int)(WEL / 8));
  cvt_f32_bf16<<<(int)((WEL / 8 + 255) / 256), 256, 0, stream>>>(Wp, wpb, (int)(WEL / 8));

  // Q/K/V projections
  dim3 gG(NEMB / 64, MROWS / 128);
  proj_gemm<<<gG, 256, 0, stream>>>(xb, wqb, (void*)qb,  bp, 0);
  proj_gemm<<<gG, 256, 0, stream>>>(xb, wkb, (void*)kbf, bp, 0);
  proj_gemm<<<gG, 256, 0, stream>>>(xb, wvb, (void*)vtb, bp, 1);

  // attention: 64 (b,h) * 128 q-tiles / 8 waves-per-block
  attn_kernel<<<(BATCH * NHEAD * (SEQ / 16)) / 8, 256, 0, stream>>>(qb, kbf, vtb, ab);

  // output projection + bias (f32)
  proj_gemm<<<gG, 256, 0, stream>>>(ab, wpb, d_out, bp, 2);
}
